// MTILayer_73976516706888
// MI455X (gfx1250) — matrix operand order repaired and checked by reading, not compiled, not run on hardware
//
#include <hip/hip_runtime.h>

// ---------------- CDNA5 WMMA types ----------------
typedef __attribute__((ext_vector_type(16))) __bf16 v16bf;
typedef __attribute__((ext_vector_type(8)))  float  v8f;

#define N_DST   20000
#define DEG     16
#define EMB     256
#define NTYPES  50
#define NHEADS  5
#define HD      10
#define S_LEN   17          // DEG + 1 (mean row)
#define NPW     4           // nodes (waves) per workgroup
#define XSTR    264         // 256 + 8 ushort padding (bank-conflict avoidance)
#define NT_TILES 4          // 50 -> 64 cols, 4 N-tiles of 16
#define KB_TILES 8          // 256 / 32
#define TILES    (NT_TILES * KB_TILES)             // 32 linearized (nt,kb) steps
#define WB_ELTS  (TILES * 32 * 16)                 // 16384 ushorts = 32 KB

union V16U { uint4 q[2]; v16bf v; };

__device__ __forceinline__ unsigned short f2bf(float f) {
    unsigned u = __float_as_uint(f);
    unsigned r = u + 0x7FFFu + ((u >> 16) & 1u);   // round-to-nearest-even
    return (unsigned short)(r >> 16);
}

__device__ __forceinline__ void store_relu_bf4(unsigned short* p, float4 v) {
    unsigned short h0 = f2bf(fmaxf(v.x, 0.0f));
    unsigned short h1 = f2bf(fmaxf(v.y, 0.0f));
    unsigned short h2 = f2bf(fmaxf(v.z, 0.0f));
    unsigned short h3 = f2bf(fmaxf(v.w, 0.0f));
    uint2 u;
    u.x = (unsigned)h0 | ((unsigned)h1 << 16);
    u.y = (unsigned)h2 | ((unsigned)h3 << 16);
    *(uint2*)p = u;
}

__device__ __forceinline__ float4 f4add(float4 a, float4 b) {
    return make_float4(a.x + b.x, a.y + b.y, a.z + b.z, a.w + b.w);
}

// -------- prep: swizzle W_fc [256x50] f32 -> WMMA-B bf16 tiles in d_ws --------
// Layout: tile t = nt*8+kb ; per tile, per lane L: n = L&15, half = L>>4,
// elem i -> K = kb*32 + half*16 + i, col = nt*16+n.  16 contiguous bf16 / lane.
__global__ void prep_w_kernel(const float* __restrict__ W, unsigned short* __restrict__ wsW) {
    int idx = blockIdx.x * blockDim.x + threadIdx.x;     // 1024 = 32 tiles * 32 lanes
    if (idx >= TILES * 32) return;
    int tile = idx >> 5;
    int lane = idx & 31;
    int nt = tile >> 3, kb = tile & 7;
    int n = lane & 15, half = lane >> 4;
    int col = nt * 16 + n;
    unsigned short* dst = wsW + ((size_t)tile * 32 + lane) * 16;
#pragma unroll
    for (int i = 0; i < 16; ++i) {
        int K = kb * 32 + half * 16 + i;
        float v = (col < NTYPES) ? W[K * NTYPES + col] : 0.0f;
        dst[i] = f2bf(v);
    }
}

// ---------------- fused main kernel: 1 wave = 1 node ----------------
// LDS is ~87 KB/WG -> 3 workgroups per WGP; request matching VGPR budget
// (3 waves/SIMD) so the allocator has slack and never spills to scratch.
__global__ void __launch_bounds__(128, 3)
mti_fused_kernel(const float* __restrict__ srcE,
                 const float* __restrict__ edgeE,
                 const float* __restrict__ aggE,
                 const int*   __restrict__ esrc,
                 const float* __restrict__ bfc,
                 const unsigned short* __restrict__ wsW,
                 float* __restrict__ out) {
    __shared__ unsigned short Xs[NPW][S_LEN][XSTR];   // relu'd bf16 rows
    __shared__ unsigned short Wb[WB_ELTS];            // swizzled bf16 W (32 KB)
    __shared__ float pred[NPW][S_LEN][52];            // FC output [17][50]
    __shared__ float scr[NPW][S_LEN][20];             // softmax rows
    __shared__ float wrow[NPW][20];                   // attn column means
    __shared__ float outb[NPW][52];                   // blended result

    const int tid  = threadIdx.x;
    const int w    = tid >> 5;            // wave / node-in-group
    const int lane = tid & 31;
    const int node = blockIdx.x * NPW + w;
    const int m    = lane & 15;
    const int half = lane >> 4;
    const float scale = 0.31622776601683794f;   // 1/sqrt(10)

    // cooperative copy of swizzled W: global -> LDS (uint4 granules, 32 KB)
    {
        const uint4* src4 = (const uint4*)wsW;
        uint4* dst4 = (uint4*)Wb;
#pragma unroll
        for (int i = 0; i < (WB_ELTS / 8) / 128; ++i)   // 2048 uint4 / 128 thr = 16
            dst4[tid + i * 128] = src4[tid + i * 128];
    }

    // zero output accumulator
    for (int i = tid; i < NPW * 52; i += blockDim.x) ((float*)outb)[i] = 0.0f;

    // per-column bias, hoisted once (nt-indexed, stream-invariant)
    float biasv[NT_TILES];
#pragma unroll
    for (int nt = 0; nt < NT_TILES; ++nt) {
        int c = nt * 16 + m;
        biasv[nt] = (c < NTYPES) ? bfc[c] : 0.0f;
    }

    // edge source indices for this node (one per lane 0..15)
    int myE = (lane < DEG) ? esrc[node * DEG + lane] : 0;

    for (int s = 0; s < 2; ++s) {
        const float* base = (s == 0) ? srcE : aggE;
        const float wfac  = 0.5f;   // BETA = 0.5 both streams

        // ---- Phase 1: gather + add, relu->bf16 rows, raw mean row ----
        float4 acc0 = make_float4(0, 0, 0, 0);
        float4 acc1 = make_float4(0, 0, 0, 0);
#pragma unroll
        for (int e = 0; e < DEG; ++e) {
            int si = __shfl(myE, e, 32);
            const float4* ar = (const float4*)(base  + (size_t)si * EMB);
            const float4* br = (const float4*)(edgeE + ((size_t)node * DEG + e) * EMB);
            float4 v0 = f4add(ar[lane],      br[lane]);
            float4 v1 = f4add(ar[lane + 32], br[lane + 32]);
            acc0 = f4add(acc0, v0);
            acc1 = f4add(acc1, v1);
            store_relu_bf4(&Xs[w][e][lane * 4],       v0);
            store_relu_bf4(&Xs[w][e][128 + lane * 4], v1);
        }
        const float inv16 = 1.0f / 16.0f;
        float4 m0 = make_float4(acc0.x * inv16, acc0.y * inv16, acc0.z * inv16, acc0.w * inv16);
        float4 m1 = make_float4(acc1.x * inv16, acc1.y * inv16, acc1.z * inv16, acc1.w * inv16);
        store_relu_bf4(&Xs[w][DEG][lane * 4],       m0);   // relu(mean) row 16
        store_relu_bf4(&Xs[w][DEG][128 + lane * 4], m1);
        __syncthreads();   // Xs (and, first iter, Wb) visible to all waves

        // ---- Phase 2a: edge-tile GEMM [16x256] @ [256x64] via bf16 WMMA ----
        {
            // A tiles (nt-invariant): 8 x v16bf = 64 VGPRs, loaded once
            V16U ua[KB_TILES];
#pragma unroll
            for (int kb = 0; kb < KB_TILES; ++kb) {
                int k0 = kb * 32 + half * 16;
                ua[kb].q[0] = *(const uint4*)&Xs[w][m][k0];        // K = k0..k0+7
                ua[kb].q[1] = *(const uint4*)&Xs[w][m][k0 + 8];    // K = k0+8..k0+15
            }
            v8f acc[NT_TILES];
#pragma unroll
            for (int nt = 0; nt < NT_TILES; ++nt) acc[nt] = (v8f){0, 0, 0, 0, 0, 0, 0, 0};

            // linearized (nt,kb) sequence with 1-ahead B prefetch from LDS
            const uint4* bp0 = (const uint4*)&Wb[(size_t)lane * 16];   // tile 0, this lane
            V16U ub[2];
            ub[0].q[0] = bp0[0];
            ub[0].q[1] = bp0[1];
#pragma unroll
            for (int t = 0; t < TILES; ++t) {
                if (t + 1 < TILES) {
                    const uint4* bp = (const uint4*)&Wb[((size_t)(t + 1) * 32 + lane) * 16];
                    ub[(t + 1) & 1].q[0] = bp[0];
                    ub[(t + 1) & 1].q[1] = bp[1];
                }
                acc[t >> 3] = __builtin_amdgcn_wmma_f32_16x16x32_bf16(
                    false, ua[t & 7].v, false, ub[t & 1].v, (short)0, acc[t >> 3], false, false);
            }
#pragma unroll
            for (int nt = 0; nt < NT_TILES; ++nt) {
                int c = nt * 16 + m;
                if (c < NTYPES) {
#pragma unroll
                    for (int v = 0; v < 8; ++v) {
                        int row = v + 8 * half;                // edge row 0..15
                        pred[w][row][c] = acc[nt][v] + biasv[nt];
                    }
                }
            }
        }

        // ---- Phase 2b: wave 0 runs the mean-row tile for all NPW nodes ----
        if (w == 0) {
            V16U ua[KB_TILES];
#pragma unroll
            for (int kb = 0; kb < KB_TILES; ++kb) {
                int k0 = kb * 32 + half * 16;
                if (m < NPW) {
                    ua[kb].q[0] = *(const uint4*)&Xs[m][DEG][k0];
                    ua[kb].q[1] = *(const uint4*)&Xs[m][DEG][k0 + 8];
                } else {
                    ua[kb].q[0] = make_uint4(0, 0, 0, 0);
                    ua[kb].q[1] = make_uint4(0, 0, 0, 0);
                }
            }
            v8f acc[NT_TILES];
#pragma unroll
            for (int nt = 0; nt < NT_TILES; ++nt) acc[nt] = (v8f){0, 0, 0, 0, 0, 0, 0, 0};

            const uint4* bp0 = (const uint4*)&Wb[(size_t)lane * 16];
            V16U ub[2];
            ub[0].q[0] = bp0[0];
            ub[0].q[1] = bp0[1];
#pragma unroll
            for (int t = 0; t < TILES; ++t) {
                if (t + 1 < TILES) {
                    const uint4* bp = (const uint4*)&Wb[((size_t)(t + 1) * 32 + lane) * 16];
                    ub[(t + 1) & 1].q[0] = bp[0];
                    ub[(t + 1) & 1].q[1] = bp[1];
                }
                acc[t >> 3] = __builtin_amdgcn_wmma_f32_16x16x32_bf16(
                    false, ua[t & 7].v, false, ub[t & 1].v, (short)0, acc[t >> 3], false, false);
            }
#pragma unroll
            for (int nt = 0; nt < NT_TILES; ++nt) {
                int c = nt * 16 + m;
                if (c < NTYPES) {
#pragma unroll
                    for (int v = 0; v < 8; ++v) {
                        int row = v + 8 * half;            // = node index in group
                        if (row < NPW) pred[row][DEG][c] = acc[nt][v] + biasv[nt];
                    }
                }
            }
        }
        __syncthreads();

        // ---- Phase 3: 5-head self-attention over [17][50], mean-pooled ----
        for (int h = 0; h < NHEADS; ++h) {
            const int cb = h * HD;
            if (lane < S_LEN) {
                float xr[HD];
#pragma unroll
                for (int d = 0; d < HD; ++d) xr[d] = pred[w][lane][cb + d];
                float att[S_LEN];
                float mx = -1e30f;
#pragma unroll
                for (int t = 0; t < S_LEN; ++t) {
                    float sc = 0.0f;
#pragma unroll
                    for (int d = 0; d < HD; ++d) sc += xr[d] * pred[w][t][cb + d];
                    sc *= scale;
                    att[t] = sc;
                    mx = fmaxf(mx, sc);
                }
                float sum = 0.0f;
#pragma unroll
                for (int t = 0; t < S_LEN; ++t) { att[t] = __expf(att[t] - mx); sum += att[t]; }
                float inv = 1.0f / sum;
#pragma unroll
                for (int t = 0; t < S_LEN; ++t) scr[w][lane][t] = att[t] * inv;
            }
            __syncthreads();
            // column means of attn (mean over query axis s)
            if (lane < S_LEN) {
                float cs = 0.0f;
#pragma unroll
                for (int i = 0; i < S_LEN; ++i) cs += scr[w][i][lane];
                wrow[w][lane] = cs * (1.0f / 17.0f);
            }
            __syncthreads();
            // result[cb+d] = sum_t wrow[t] * x[t][cb+d]
            if (lane < HD) {
                float r = 0.0f;
#pragma unroll
                for (int t = 0; t < S_LEN; ++t) r += wrow[w][t] * pred[w][t][cb + lane];
                outb[w][cb + lane] += wfac * r;
            }
            __syncthreads();
        }
    }

    // ---- write final [NTYPES] per node ----
    for (int t = lane; t < NTYPES; t += 32)
        out[(size_t)node * NTYPES + t] = outb[w][t];
}

extern "C" void kernel_launch(void* const* d_in, const int* in_sizes, int n_in,
                              void* d_out, int out_size, void* d_ws, size_t ws_size,
                              hipStream_t stream) {
    const float* srcE  = (const float*)d_in[0];   // [20000,256]
    const float* edgeE = (const float*)d_in[1];   // [320000,256]
    const float* aggE  = (const float*)d_in[2];   // [20000,256]
    const int*   esrc  = (const int*)  d_in[3];   // [320000]
    // d_in[4] = edge_dst (implicit: DEG per node, sorted) — unused
    const float* Wfc   = (const float*)d_in[5];   // [256,50]
    const float* bfc   = (const float*)d_in[6];   // [50]
    float* out = (float*)d_out;
    unsigned short* wsW = (unsigned short*)d_ws;  // 32 KB swizzled bf16 W

    prep_w_kernel<<<4, 256, 0, stream>>>(Wfc, wsW);
    mti_fused_kernel<<<N_DST / NPW, 128, 0, stream>>>(srcE, edgeE, aggE, esrc, bfc, wsW, out);
}
